// MultiHeadAttention_25314537242916
// MI455X (gfx1250) — matrix operand order repaired and checked by reading, not compiled, not run on hardware
//
#include <hip/hip_runtime.h>
#include <hip/hip_bf16.h>
#include <stdint.h>

// ---------------------------------------------------------------------------
// MI455X (gfx1250) MHA pipeline, bf16 WMMA, f32 accumulate.
//   K0: weight f32->bf16 (Wq/Wk/Wv concatenated -> one [3072][1024] matrix)
//   K1: QKV = h @ Wqkv^T          (f32 in, bf16 out)   - tiled WMMA GEMM
//   K2: per-node 16x16 attention  (WMMA + wave32 shuffle softmax)
//   K3: out = A @ Wfc^T           (bf16 in, f32 out)   - tiled WMMA GEMM
// Memory-bound problem (~1.5 GB HBM traffic @ 23.3 TB/s); bf16 halves all
// intermediate traffic while WMMA gives f32 accumulation.
// ---------------------------------------------------------------------------

typedef __attribute__((ext_vector_type(16))) __bf16 v16bf;
typedef __attribute__((ext_vector_type(8)))  __bf16 v8bf;
typedef __attribute__((ext_vector_type(8)))  float  v8f;
typedef int v4i_vec __attribute__((vector_size(16)));   // 4 x i32

#define NODES   65536
#define HID     1024
#define QKVCOLS 3072

union V16U { v16bf v; v8bf h[2]; };

static __device__ inline v16bf mk16(v8bf lo, v8bf hi) {
  V16U u; u.h[0] = lo; u.h[1] = hi; return u.v;
}
static __device__ inline v8bf bzero8() {
  v8bf z;
  #pragma unroll
  for (int e = 0; e < 8; ++e) z[e] = (__bf16)0.0f;
  return z;
}
static __device__ inline v8f fzero8() {
  v8f z;
  #pragma unroll
  for (int e = 0; e < 8; ++e) z[e] = 0.0f;
  return z;
}

// ---- CDNA5 async global->LDS (guarded; falls back to register staging) ----
#if __has_builtin(__builtin_amdgcn_global_load_async_to_lds_b128)
#define HAVE_ASYNC_LDS 1
typedef __attribute__((address_space(1))) v4i_vec* g_v4i_ptr;
typedef __attribute__((address_space(3))) v4i_vec* l_v4i_ptr;
static __device__ inline void async_cp16(const __bf16* g, __bf16* l) {
  __builtin_amdgcn_global_load_async_to_lds_b128(
      (g_v4i_ptr)(void*)g, (l_v4i_ptr)(void*)l, 0, 0);
}
static __device__ inline void wait_async0() {
#if __has_builtin(__builtin_amdgcn_s_wait_asynccnt)
  __builtin_amdgcn_s_wait_asynccnt(0);
#else
  asm volatile("s_wait_asynccnt 0" ::: "memory");
#endif
}
#else
#define HAVE_ASYNC_LDS 0
#endif

// ---- staging helpers ----
template <typename T>
static __device__ inline void fetch16(const T* __restrict__ g, T* r) {
  #pragma unroll
  for (int e = 0; e < 16; ++e) r[e] = g[e];
}
static __device__ inline void store16(const float* r, __bf16* dst) {
  v8bf b0, b1;
  #pragma unroll
  for (int e = 0; e < 8; ++e) { b0[e] = (__bf16)r[e]; b1[e] = (__bf16)r[8 + e]; }
  *(v8bf*)dst       = b0;
  *(v8bf*)(dst + 8) = b1;
}
static __device__ inline void store16(const __bf16* r, __bf16* dst) {
  *(v8bf*)dst       = *(const v8bf*)r;
  *(v8bf*)(dst + 8) = *(const v8bf*)(r + 8);
}
static __device__ inline void storeC(float* C, size_t idx, float v)  { C[idx] = v; }
static __device__ inline void storeC(__bf16* C, size_t idx, float v) { C[idx] = (__bf16)v; }

// ---------------------------------------------------------------------------
// K0: weight conversion.
// ---------------------------------------------------------------------------
__global__ __launch_bounds__(256)
void convert_weights(const float* __restrict__ Wq, const float* __restrict__ Wk,
                     const float* __restrict__ Wv, const float* __restrict__ Wfc,
                     __bf16* __restrict__ Wqkv, __bf16* __restrict__ Wfcb) {
  const size_t MM = 1024ull * 1024ull;
  const size_t i = (size_t)blockIdx.x * 256 + threadIdx.x;
  if (i < MM)               Wqkv[i] = (__bf16)Wq[i];
  else if (i < 2 * MM)      Wqkv[i] = (__bf16)Wk[i - MM];
  else if (i < 3 * MM)      Wqkv[i] = (__bf16)Wv[i - 2 * MM];
  else                      Wfcb[i - 3 * MM] = (__bf16)Wfc[i - 3 * MM];
}

// ---------------------------------------------------------------------------
// Tiled GEMM: C[M][N] = X[M][K] @ W[N][K]^T.
// 128x256 block tile, 256 threads = 8 waves (4 down x 2 across),
// each wave owns 32x128 = 2x8 WMMA tiles. Double-buffered LDS; weight
// stream staged via CDNA5 async global->LDS when available.
// ---------------------------------------------------------------------------
template <typename TIN, typename TOUT>
__global__ __launch_bounds__(256)
void gemm_xwt(const TIN* __restrict__ X, const __bf16* __restrict__ W,
              TOUT* __restrict__ C, int M, int K, int N) {
  constexpr int BM = 128, BN = 256, BK = 32, LDT = 40;  // 40 = 32 + 8 pad
  __shared__ __bf16 As[2][BM * LDT];
  __shared__ __bf16 Bs[2][BN * LDT];

  const int rowBase = blockIdx.y * BM;
  const int colBase = blockIdx.x * BN;
  const int t     = threadIdx.x;
  const int lane  = t & 31;
  const int wv    = t >> 5;
  const int waveM = (wv & 3) * 32;
  const int waveN = (wv >> 2) * 128;
  const int half  = lane >> 4;   // wave32 half-select per ISA A/B/C layouts
  const int l16   = lane & 15;

  // A staging: thread t covers row (t>>1), 16 cols at (t&1)*16.
  const int ar = t >> 1;
  const int ac = (t & 1) * 16;
  const TIN*    aptr = X + (size_t)(rowBase + ar) * K + ac;
  // B staging: thread t covers row t (one weight row), 32 cols.
  const __bf16* bptr = W + (size_t)(colBase + t) * K;

  v8f acc[2][8];
  #pragma unroll
  for (int i = 0; i < 2; ++i)
    #pragma unroll
    for (int j = 0; j < 8; ++j) acc[i][j] = fzero8();

  const int NK = K / BK;
  TIN aReg[16];
#if !HAVE_ASYNC_LDS
  __bf16 bReg[32];
#endif

  // -------- prologue: fill buffer 0 --------
  fetch16(aptr, aReg);
  store16(aReg, &As[0][ar * LDT + ac]);
#if HAVE_ASYNC_LDS
  #pragma unroll
  for (int c = 0; c < 4; ++c)
    async_cp16(bptr + c * 8, &Bs[0][t * LDT + c * 8]);
#else
  fetch16(bptr, bReg); fetch16(bptr + 16, bReg + 16);
  store16(bReg, &Bs[0][t * LDT]); store16(bReg + 16, &Bs[0][t * LDT + 16]);
#endif

  for (int kt = 0; kt < NK; ++kt) {
    const int cur = kt & 1;
    const int nxt = cur ^ 1;
#if HAVE_ASYNC_LDS
    wait_async0();
#endif
    __syncthreads();

    const int k1 = (kt + 1) * BK;
    if (kt + 1 < NK) {
      fetch16(aptr + k1, aReg);
#if HAVE_ASYNC_LDS
      #pragma unroll
      for (int c = 0; c < 4; ++c)
        async_cp16(bptr + k1 + c * 8, &Bs[nxt][t * LDT + c * 8]);
#else
      fetch16(bptr + k1, bReg); fetch16(bptr + k1 + 16, bReg + 16);
#endif
      __builtin_prefetch(bptr + k1 + BK, 0, 0);   // global_prefetch_b8
    }

    // ---- compute on buffer `cur` ----
    // A frag: lane<16 -> K {0..7,16..23}, lane>=16 -> K {8..15,24..31}.
    v16bf afr[2];
    #pragma unroll
    for (int i = 0; i < 2; ++i) {
      const __bf16* ap = &As[cur][(waveM + i * 16 + l16) * LDT];
      afr[i] = mk16(*(const v8bf*)(ap + half * 8),
                    *(const v8bf*)(ap + 16 + half * 8));
    }
    // B frag: lane = out col, lane<16 -> K {0..7,16..23}, lane>=16 -> K {8..15,24..31}.
    #pragma unroll
    for (int j = 0; j < 8; ++j) {
      const __bf16* bp = &Bs[cur][(waveN + j * 16 + l16) * LDT];
      const v16bf bfr = mk16(*(const v8bf*)(bp + half * 8), *(const v8bf*)(bp + 16 + half * 8));
      #pragma unroll
      for (int i = 0; i < 2; ++i)
        acc[i][j] = __builtin_amdgcn_wmma_f32_16x16x32_bf16(
            false, afr[i], false, bfr, (short)0, acc[i][j], false, false);
    }

    if (kt + 1 < NK) {
      store16(aReg, &As[nxt][ar * LDT + ac]);
#if !HAVE_ASYNC_LDS
      store16(bReg, &Bs[nxt][t * LDT]);
      store16(bReg + 16, &Bs[nxt][t * LDT + 16]);
#endif
    }
  }

  // C layout: lane = N, VGPR vv -> M = vv + 8*(lane>=16).
  #pragma unroll
  for (int i = 0; i < 2; ++i)
    #pragma unroll
    for (int j = 0; j < 8; ++j) {
      const int col = colBase + waveN + j * 16 + l16;
      #pragma unroll
      for (int vv = 0; vv < 8; ++vv) {
        const int row = rowBase + waveM + i * 16 + vv + 8 * half;
        storeC(C, (size_t)row * N + col, acc[i][j][vv]);
      }
    }
}

// ---------------------------------------------------------------------------
// K2: per-node attention, one wave per node.
//   comp = Q(16x64) @ Kraw(64x16) / 128 ; softmax rows ; out = S @ Vraw^T ;
//   stored transposed (d,l). K is staged TRANSPOSED into LDS so the WMMA
//   B-fragment becomes two aligned ds_load_b128 instead of 16 ds_load_u16.
// ---------------------------------------------------------------------------
__global__ __launch_bounds__(256)
void attn_kernel(const __bf16* __restrict__ qkv, __bf16* __restrict__ aout) {
  constexpr int KT = 72;            // transposed-K row stride (64 + 8 pad)
  __shared__ __bf16 kT[8][16 * KT]; // kT[g][k] = Krow[k*16+g]
  __shared__ __bf16 vr[8][1024];    // V row, raw
  __shared__ __bf16 sb[8][256];     // 16x16 softmax scores

  const int t    = threadIdx.x;
  const int lane = t & 31;
  const int wv   = t >> 5;
  const int half = lane >> 4;
  const int l16  = lane & 15;
  const size_t node = (size_t)blockIdx.x * 8 + wv;
  const __bf16* base = qkv + node * (size_t)QKVCOLS;

  // Stage K (transposed scatter) and V (straight copy) into LDS.
  #pragma unroll
  for (int i = 0; i < 4; ++i) {
    const int idx = i * 256 + lane * 8;
    const v8bf kx = *(const v8bf*)(base + 1024 + idx);
    #pragma unroll
    for (int e = 0; e < 8; ++e) {
      const int q = idx + e;                    // q = k*16 + g
      kT[wv][(q & 15) * KT + (q >> 4)] = kx[e];
    }
    *(v8bf*)&vr[wv][idx] = *(const v8bf*)(base + 2048 + idx);
  }
  __syncthreads();

  // comp = Q @ Kraw : two K=32 WMMAs.
  v8f acc = fzero8();
  #pragma unroll
  for (int c = 0; c < 2; ++c) {
    const __bf16* qp = base + l16 * 64 + c * 32;
    const v16bf a = mk16(*(const v8bf*)(qp + half * 16),
                         *(const v8bf*)(qp + half * 16 + 8));
    const __bf16* kp = &kT[wv][l16 * KT + c * 32 + half * 16];
    const v16bf b = mk16(*(const v8bf*)kp, *(const v8bf*)(kp + 8));
    acc = __builtin_amdgcn_wmma_f32_16x16x32_bf16(false, a, false, b,
                                                  (short)0, acc, false, false);
  }

  // Softmax over last axis (columns live across each 16-lane half).
  const float scale = 1.0f / 128.0f;            // (1/head_dim) * 0.5
  float s[8];
  #pragma unroll
  for (int vvv = 0; vvv < 8; ++vvv) {
    float x = acc[vvv] * scale;
    float m = x;
    #pragma unroll
    for (int d = 1; d < 16; d <<= 1) m = fmaxf(m, __shfl_xor(m, d, 32));
    const float e = __expf(x - m);
    float sum = e;
    #pragma unroll
    for (int d = 1; d < 16; d <<= 1) sum += __shfl_xor(sum, d, 32);
    s[vvv] = e / sum;
  }
  #pragma unroll
  for (int vvv = 0; vvv < 8; ++vvv)
    sb[wv][(vvv + 8 * half) * 16 + l16] = (__bf16)s[vvv];   // S[M][g]
  __syncthreads();

  // out = S(16x16, zero-padded to K=32) @ Vraw^T, 4 chunks of 16 cols.
  const v8bf z8 = bzero8();
  const v16bf sA = half == 0 ? mk16(*(const v8bf*)&sb[wv][l16 * 16], *(const v8bf*)&sb[wv][l16 * 16 + 8]) : mk16(z8, z8);
  #pragma unroll
  for (int dc = 0; dc < 4; ++dc) {
    v16bf bV;
    if (half == 0) {   // valid k = 0..15 only; upper half hits zeroed A
      const __bf16* vp = &vr[wv][(dc * 16 + l16) * 16];
      bV = mk16(*(const v8bf*)vp, *(const v8bf*)(vp + 8));
    } else {
      bV = mk16(z8, z8);
    }
    v8f o = fzero8();
    o = __builtin_amdgcn_wmma_f32_16x16x32_bf16(false, sA, false, bV,
                                                (short)0, o, false, false);
    // Transposed store: A[n][d*16 + l], d = dc*16 + N, l = vv + 8*half.
    #pragma unroll
    for (int vvv = 0; vvv < 8; ++vvv)
      aout[node * 1024 + (size_t)(dc * 16 + l16) * 16 + vvv + 8 * half] =
          (__bf16)o[vvv];
  }
}

// ---------------------------------------------------------------------------
extern "C" void kernel_launch(void* const* d_in, const int* in_sizes, int n_in,
                              void* d_out, int out_size, void* d_ws, size_t ws_size,
                              hipStream_t stream) {
  (void)in_sizes; (void)n_in; (void)out_size; (void)ws_size;
  const float* h   = (const float*)d_in[0];
  const float* Wq  = (const float*)d_in[1];
  const float* Wk  = (const float*)d_in[2];
  const float* Wv  = (const float*)d_in[3];
  const float* Wfc = (const float*)d_in[4];
  float* out = (float*)d_out;

  // Workspace (bf16): Wqkv 6MB | Wfc 2MB | QKV 384MB | A 128MB
  __bf16* Wqkv = (__bf16*)d_ws;
  __bf16* Wfcb = Wqkv + 3ull * 1024 * 1024;
  __bf16* QKV  = Wfcb + 1ull * 1024 * 1024;
  __bf16* Aout = QKV  + (size_t)NODES * QKVCOLS;

  convert_weights<<<(4 * 1024 * 1024) / 256, 256, 0, stream>>>(
      Wq, Wk, Wv, Wfc, Wqkv, Wfcb);

  gemm_xwt<float, __bf16><<<dim3(QKVCOLS / 256, NODES / 128), 256, 0, stream>>>(
      h, Wqkv, QKV, NODES, HID, QKVCOLS);

  attn_kernel<<<NODES / 8, 256, 0, stream>>>(QKV, Aout);

  gemm_xwt<__bf16, float><<<dim3(HID / 256, NODES / 128), 256, 0, stream>>>(
      Aout, Wfcb, out, NODES, HID, HID);
}
